// DeformableCrossAttentionBlock_5111011082602
// MI455X (gfx1250) — matrix operand order repaired and checked by reading, not compiled, not run on hardware
//
#include <hip/hip_runtime.h>
#include <hip/hip_bf16.h>
#include <math.h>

// ---------------- problem constants ----------------
#define BSZ   16
#define SEQ   1024
#define DIM   1024
#define NHEAD 16
#define KPTS  4
#define HDIM  64
#define HS    32
#define WSZ   32
#define ROWS  (BSZ * SEQ)      // 16384
#define D4    (4 * DIM)        // 4096

typedef __attribute__((ext_vector_type(16))) _Float16 v16h;
typedef __attribute__((ext_vector_type(8)))  _Float16 v8h;
typedef __attribute__((ext_vector_type(8)))  float    v8f;
typedef __attribute__((ext_vector_type(4)))  unsigned u32x4;
typedef __attribute__((ext_vector_type(8)))  unsigned u32x8;

__device__ __forceinline__ float gelu_exact(float x) {
    return 0.5f * x * (1.0f + erff(x * 0.70710678118654752f));
}

// set a bitfield inside a little-endian packed dword array (for TDM D# build)
__device__ __forceinline__ void set_bits(unsigned* a, int bit, int width,
                                         unsigned long long val) {
    #pragma unroll
    for (int i = 0; i < 48; ++i) {
        if (i >= width) break;
        const int b = bit + i;
        a[b >> 5] |= (unsigned)((val >> i) & 1ull) << (b & 31);
    }
}

// ---------------- f32 -> f16 convert ----------------
__global__ __launch_bounds__(256) void cvt_f32_f16(const float* __restrict__ src,
                                                   _Float16* __restrict__ dst, int n) {
    int i = blockIdx.x * 256 + threadIdx.x;
    if (i < n) dst[i] = (_Float16)src[i];
}

// ---------------- LayerNorm (one block per row of DIM) ----------------
__global__ __launch_bounds__(256) void layernorm_f16(const float* __restrict__ x,
                                                     const float* __restrict__ g,
                                                     const float* __restrict__ b,
                                                     _Float16* __restrict__ out) {
    __shared__ float rs[256], rq[256];
    const int row = blockIdx.x;
    const int tid = threadIdx.x;
    const float* xr = x + (size_t)row * DIM;
    float4 v = *(const float4*)(xr + tid * 4);
    float s  = v.x + v.y + v.z + v.w;
    float sq = v.x * v.x + v.y * v.y + v.z * v.z + v.w * v.w;
    rs[tid] = s; rq[tid] = sq;
    __syncthreads();
    for (int off = 128; off > 0; off >>= 1) {
        if (tid < off) { rs[tid] += rs[tid + off]; rq[tid] += rq[tid + off]; }
        __syncthreads();
    }
    const float mean = rs[0] * (1.0f / DIM);
    const float var  = rq[0] * (1.0f / DIM) - mean * mean;
    const float rstd = rsqrtf(var + 1e-5f);
    _Float16* orow = out + (size_t)row * DIM;
    const int i = tid * 4;
    float4 gg = *(const float4*)(g + i);
    float4 bb = *(const float4*)(b + i);
    orow[i + 0] = (_Float16)((v.x - mean) * rstd * gg.x + bb.x);
    orow[i + 1] = (_Float16)((v.y - mean) * rstd * gg.y + bb.y);
    orow[i + 2] = (_Float16)((v.z - mean) * rstd * gg.z + bb.z);
    orow[i + 3] = (_Float16)((v.w - mean) * rstd * gg.w + bb.w);
}

// ---------------- WMMA GEMM: C = act(A(f16)[M,K] @ B(f16)[K,N] + bias + resid) ----------------
// BM=128, BN=128, BK=32; 8 wave32s; each wave: 32x64 = 2x4 v_wmma_f32_16x16x32_f16 tiles.
// A tile is staged global->LDS by the Tensor Data Mover (one descriptor per k-step,
// with LDS padding 4 dwords per 16 dwords to produce the 40-half padded stride);
// B tile is staged via VMEM + transposing ds_store so K is contiguous per column.
template<int ACT, bool HALF_OUT>
__global__ __launch_bounds__(256) void gemm_f16_wmma(const _Float16* __restrict__ A,
                                                     const _Float16* __restrict__ Bw,
                                                     const float* __restrict__ bias,
                                                     const float* __restrict__ resid,
                                                     float* __restrict__ Cf,
                                                     _Float16* __restrict__ Ch,
                                                     int M, int N, int Kd) {
    constexpr int BM = 128, BN = 128, BK = 32;
    constexpr int LDA = BK + 8;   // 40 halves (80B rows -> 16B aligned subreads)
    constexpr int LDB = BK + 8;
    __shared__ __align__(16) _Float16 sA[BM * LDA];
    __shared__ __align__(16) _Float16 sB[BN * LDB];

    const int tid  = threadIdx.x;
    const int wave = tid >> 5;
    const int lane = tid & 31;
    const int wm   = wave >> 1;          // 0..3 -> 32-row strip
    const int wn   = wave & 1;           // 0..1 -> 64-col strip
    const int lr   = lane & 15;          // lane row/col within 16
    const int hf   = lane >> 4;          // half-wave select
    const int blockM = blockIdx.y * BM;
    const int blockN = blockIdx.x * BN;

    v8f acc[2][4];
    {
        v8f zero = {0.f, 0.f, 0.f, 0.f, 0.f, 0.f, 0.f, 0.f};
        #pragma unroll
        for (int mi = 0; mi < 2; ++mi)
            #pragma unroll
            for (int ni = 0; ni < 4; ++ni) acc[mi][ni] = zero;
    }

    const int bK  = tid >> 3;            // 0..31  (K row for B staging)
    const int bCg = (tid & 7) * 16;      // 0..112 (col group, 16 cols/thread)

    // uniform pieces of the TDM descriptor
    const unsigned ldsA = (unsigned)(size_t)(&sA[0]);   // wave-relative LDS offset
    const unsigned long long gbase =
        (unsigned long long)(size_t)(A + (size_t)blockM * Kd);

    for (int kb = 0; kb < Kd; kb += BK) {
        // --- stage A tile (128 x 32 halves) via Tensor Data Mover, wave 0 only ---
        if (wave == 0) {
            const unsigned long long ga = gbase + (unsigned long long)kb * 2ull;
            unsigned g0[4] = {0u, 0u, 0u, 0u};
            g0[0] = 1u;                         // count = 1 valid descriptor
            g0[1] = ldsA;                       // lds_addr
            g0[2] = (unsigned)ga;               // global_addr[31:0]
            g0[3] = (unsigned)(ga >> 32) | (2u << 30);  // global_addr[56:32] | type=2
            unsigned g1[8] = {0u, 0u, 0u, 0u, 0u, 0u, 0u, 0u};
            set_bits(g1, 16, 2, 1ull);          // data_size = 2 bytes
            set_bits(g1, 20, 1, 1ull);          // pad_enable
            set_bits(g1, 22, 3, 3ull);          // pad_interval: 16 dwords (one 64B row)
            set_bits(g1, 25, 7, 3ull);          // pad_amount: 4 dwords (16B pad -> LDA=40)
            set_bits(g1, 48, 32, (unsigned long long)Kd);   // tensor_dim0
            set_bits(g1, 80, 32, (unsigned long long)M);    // tensor_dim1
            set_bits(g1, 112, 16, (unsigned long long)BK);  // tile_dim0 = 32 elems
            set_bits(g1, 128, 16, (unsigned long long)BM);  // tile_dim1 = 128 rows
            set_bits(g1, 160, 48, (unsigned long long)Kd);  // tensor_dim0_stride
            u32x4 d0; u32x8 d1;
            #pragma unroll
            for (int j = 0; j < 4; ++j) d0[j] = g0[j];
            #pragma unroll
            for (int j = 0; j < 8; ++j) d1[j] = g1[j];
            asm volatile("tensor_load_to_lds %0, %1" :: "s"(d0), "s"(d1) : "memory");
        }
        // --- stage B tile transposed ([BN][BK]) so K is contiguous per column ---
        {
            const _Float16* src = Bw + (size_t)(kb + bK) * N + blockN + bCg;
            v8h t0 = *(const v8h*)(src);
            v8h t1 = *(const v8h*)(src + 8);
            #pragma unroll
            for (int j = 0; j < 8; ++j) {
                sB[(bCg + j) * LDB + bK]     = t0[j];
                sB[(bCg + 8 + j) * LDB + bK] = t1[j];
            }
            if (kb + BK < Kd)
                __builtin_prefetch(src + (size_t)BK * N, 0, 0);  // global_prefetch_b8
        }
        if (wave == 0)
            __builtin_amdgcn_s_wait_tensorcnt(0);   // TDM tile landed in LDS
        __syncthreads();

        // --- build fragments per documented CDNA5 wave32 layouts ---
        v16h afr[2], bfr[4];
        #pragma unroll
        for (int mi = 0; mi < 2; ++mi) {
            const _Float16* p = &sA[(wm * 32 + mi * 16 + lr) * LDA];
            v8h lo = *(const v8h*)(p + hf * 8);        // K = 8h .. 8h+7
            v8h hi = *(const v8h*)(p + 16 + hf * 8);   // K = 16+8h .. 16+8h+7
            v16h f;
            #pragma unroll
            for (int j = 0; j < 8; ++j) { f[j] = lo[j]; f[8 + j] = hi[j]; }
            afr[mi] = f;
        }
        #pragma unroll
        for (int ni = 0; ni < 4; ++ni) {
            const _Float16* p = &sB[(wn * 64 + ni * 16 + lr) * LDB];
            v8h lo = *(const v8h*)(p + hf * 8);        // K = 8h .. 8h+7
            v8h hi = *(const v8h*)(p + 16 + hf * 8);   // K = 16+8h .. 16+8h+7
            v16h f;
            #pragma unroll
            for (int j = 0; j < 8; ++j) { f[j] = lo[j]; f[8 + j] = hi[j]; }
            bfr[ni] = f;
        }
        #pragma unroll
        for (int mi = 0; mi < 2; ++mi)
            #pragma unroll
            for (int ni = 0; ni < 4; ++ni)
                acc[mi][ni] = __builtin_amdgcn_wmma_f32_16x16x32_f16(
                    false, afr[mi], false, bfr[ni], (short)0, acc[mi][ni], false, false);
        __syncthreads();
    }

    // --- epilogue: bias + residual + activation, f16 or f32 store ---
    #pragma unroll
    for (int mi = 0; mi < 2; ++mi) {
        #pragma unroll
        for (int ni = 0; ni < 4; ++ni) {
            const int col = blockN + wn * 64 + ni * 16 + lr;
            const float bv = bias ? bias[col] : 0.0f;
            #pragma unroll
            for (int v = 0; v < 8; ++v) {
                const int row = blockM + wm * 32 + mi * 16 + v + 8 * hf;
                const size_t idx = (size_t)row * N + col;
                float x = acc[mi][ni][v] + bv;
                if (resid) x += resid[idx];
                if (ACT == 1) x = gelu_exact(x);
                if (HALF_OUT) Ch[idx] = (_Float16)x;
                else          Cf[idx] = x;
            }
        }
    }
}

// ---------------- per-head offsets MLP + attention softmax ----------------
__global__ __launch_bounds__(128) void head_mlp(const _Float16* __restrict__ q,
                                                const float* __restrict__ Wo1, const float* __restrict__ bo1,
                                                const float* __restrict__ Wo2, const float* __restrict__ bo2,
                                                const float* __restrict__ Wa,  const float* __restrict__ ba,
                                                float* __restrict__ offv, float* __restrict__ attn) {
    const int t = blockIdx.x * 128 + threadIdx.x;
    if (t >= ROWS * NHEAD) return;
    const int row = t / NHEAD, n = t % NHEAD;
    float qv[HDIM];
    const _Float16* qp = q + (size_t)row * DIM + n * HDIM;
    #pragma unroll
    for (int i = 0; i < HDIM; ++i) qv[i] = (float)qp[i];

    float h[64];
    #pragma unroll 4
    for (int j = 0; j < 64; ++j) {
        float s = bo1[j];
        #pragma unroll
        for (int i = 0; i < HDIM; ++i) s += qv[i] * Wo1[i * 64 + j];
        h[j] = fmaxf(s, 0.0f);
    }
    float* op = offv + (size_t)t * (2 * KPTS);
    #pragma unroll
    for (int j = 0; j < 2 * KPTS; ++j) {
        float s = bo2[j];
        #pragma unroll
        for (int i = 0; i < 64; ++i) s += h[i] * Wo2[i * (2 * KPTS) + j];
        op[j] = tanhf(s) * 0.5f;
    }
    float lg[KPTS], mx = -3.4e38f;
    #pragma unroll
    for (int k = 0; k < KPTS; ++k) {
        float s = ba[k];
        #pragma unroll
        for (int i = 0; i < HDIM; ++i) s += qv[i] * Wa[i * KPTS + k];
        lg[k] = s; mx = fmaxf(mx, s);
    }
    float se = 0.0f;
    #pragma unroll
    for (int k = 0; k < KPTS; ++k) { lg[k] = expf(lg[k] - mx); se += lg[k]; }
    const float inv = 1.0f / se;
    float* ap = attn + (size_t)t * KPTS;
    #pragma unroll
    for (int k = 0; k < KPTS; ++k) ap[k] = lg[k] * inv;
}

// ---------------- deformable bilinear gather + attn-weighted sum ----------------
__global__ __launch_bounds__(256) void deform_gather(const float* __restrict__ vmap,
                                                     const float* __restrict__ offv,
                                                     const float* __restrict__ attn,
                                                     _Float16* __restrict__ ho) {
    const int wid  = (blockIdx.x * 256 + threadIdx.x) >> 5;
    const int lane = threadIdx.x & 31;
    const int b   = wid / (NHEAD * SEQ);
    const int rem = wid % (NHEAD * SEQ);
    const int n   = rem / SEQ;
    const int s   = rem % SEQ;
    const int row = b * SEQ + s;
    const float rx = -1.0f + 2.0f * (float)(s & (WSZ - 1)) / (float)(WSZ - 1);
    const float ry = -1.0f + 2.0f * (float)(s >> 5)        / (float)(HS - 1);
    const int d0 = lane * 2;
    const float* ob = offv + ((size_t)row * NHEAD + n) * (2 * KPTS);
    const float* ab = attn + ((size_t)row * NHEAD + n) * KPTS;
    float ax = 0.0f, ay = 0.0f;

    #pragma unroll
    for (int k = 0; k < KPTS; ++k) {
        const float a  = ab[k];
        const float px = (rx + ob[2 * k + 0] + 1.0f) * 0.5f * (float)(WSZ - 1);
        const float py = (ry + ob[2 * k + 1] + 1.0f) * 0.5f * (float)(HS - 1);
        const float xf = floorf(px), yf = floorf(py);
        const int   x0 = (int)xf,   y0 = (int)yf;
        const float wx = px - xf,   wy = py - yf;
        const float w00 = (1.f - wx) * (1.f - wy) * a, w10 = wx * (1.f - wy) * a;
        const float w01 = (1.f - wx) * wy * a,         w11 = wx * wy * a;
        #pragma unroll
        for (int c = 0; c < 4; ++c) {
            const int xi = x0 + (c & 1), yi = y0 + (c >> 1);
            const float w = (c == 0) ? w00 : (c == 1) ? w10 : (c == 2) ? w01 : w11;
            if (xi >= 0 && xi < WSZ && yi >= 0 && yi < HS) {
                const size_t src = ((size_t)(b * SEQ + yi * WSZ + xi) * DIM) + n * HDIM + d0;
                const float2 p = *(const float2*)(vmap + src);
                ax += w * p.x; ay += w * p.y;
            }
        }
    }
    _Float16* out = ho + (size_t)row * DIM + n * HDIM + d0;
    out[0] = (_Float16)ax;
    out[1] = (_Float16)ay;
}

// ---------------- host-side orchestration ----------------
extern "C" void kernel_launch(void* const* d_in, const int* in_sizes, int n_in,
                              void* d_out, int out_size, void* d_ws, size_t ws_size,
                              hipStream_t stream) {
    (void)in_sizes; (void)n_in; (void)out_size; (void)ws_size;
    const float* source = (const float*)d_in[0];
    const float* target = (const float*)d_in[1];
    const float* Wq   = (const float*)d_in[2];  const float* bq   = (const float*)d_in[3];
    const float* Wo1  = (const float*)d_in[4];  const float* bo1  = (const float*)d_in[5];
    const float* Wo2  = (const float*)d_in[6];  const float* bo2  = (const float*)d_in[7];
    const float* Wa   = (const float*)d_in[8];  const float* ba   = (const float*)d_in[9];
    const float* Wv   = (const float*)d_in[10]; const float* bv   = (const float*)d_in[11];
    const float* Wout = (const float*)d_in[12]; const float* bout = (const float*)d_in[13];
    const float* gq   = (const float*)d_in[14]; const float* bq_ln  = (const float*)d_in[15];
    const float* gkv  = (const float*)d_in[16]; const float* bkv_ln = (const float*)d_in[17];
    const float* gffn = (const float*)d_in[18]; const float* bffn_ln= (const float*)d_in[19];
    const float* W1   = (const float*)d_in[20]; const float* b1   = (const float*)d_in[21];
    const float* W2   = (const float*)d_in[22]; const float* b2   = (const float*)d_in[23];

    char* w = (char*)d_ws;
    auto alloc = [&](size_t bytes) -> void* {
        void* p = (void*)w; w += (bytes + 255) & ~(size_t)255; return p;
    };
    _Float16* qn_h   = (_Float16*)alloc((size_t)ROWS * DIM * 2);
    _Float16* vn_h   = (_Float16*)alloc((size_t)ROWS * DIM * 2);
    _Float16* q_h    = (_Float16*)alloc((size_t)ROWS * DIM * 2);
    float*    vmap   = (float*)   alloc((size_t)ROWS * DIM * 4);
    float*    offv   = (float*)   alloc((size_t)ROWS * NHEAD * 2 * KPTS * 4);
    float*    attn   = (float*)   alloc((size_t)ROWS * NHEAD * KPTS * 4);
    _Float16* ho_h   = (_Float16*)alloc((size_t)ROWS * DIM * 2);
    float*    srcres = (float*)   alloc((size_t)ROWS * DIM * 4);
    _Float16* ffn_h  = (_Float16*)alloc((size_t)ROWS * DIM * 2);
    _Float16* h1_h   = (_Float16*)alloc((size_t)ROWS * D4 * 2);
    _Float16* Wq_h   = (_Float16*)alloc((size_t)DIM * DIM * 2);
    _Float16* Wv_h   = (_Float16*)alloc((size_t)DIM * DIM * 2);
    _Float16* Wout_h = (_Float16*)alloc((size_t)DIM * DIM * 2);
    _Float16* W1_h   = (_Float16*)alloc((size_t)DIM * D4 * 2);
    _Float16* W2_h   = (_Float16*)alloc((size_t)D4 * DIM * 2);

    // weight conversions
    cvt_f32_f16<<<(DIM * DIM + 255) / 256, 256, 0, stream>>>(Wq,   Wq_h,   DIM * DIM);
    cvt_f32_f16<<<(DIM * DIM + 255) / 256, 256, 0, stream>>>(Wv,   Wv_h,   DIM * DIM);
    cvt_f32_f16<<<(DIM * DIM + 255) / 256, 256, 0, stream>>>(Wout, Wout_h, DIM * DIM);
    cvt_f32_f16<<<(DIM * D4  + 255) / 256, 256, 0, stream>>>(W1,   W1_h,   DIM * D4);
    cvt_f32_f16<<<(D4  * DIM + 255) / 256, 256, 0, stream>>>(W2,   W2_h,   D4 * DIM);

    // layernorms -> f16 GEMM inputs
    layernorm_f16<<<ROWS, 256, 0, stream>>>(source, gq,  bq_ln,  qn_h);
    layernorm_f16<<<ROWS, 256, 0, stream>>>(target, gkv, bkv_ln, vn_h);

    // q = qn @ Wq + bq (f16 out);  v = vn @ Wv + bv (f32 out, gather source)
    dim3 gD(DIM / 128, ROWS / 128), g4(D4 / 128, ROWS / 128);
    gemm_f16_wmma<0, true ><<<gD, 256, 0, stream>>>(qn_h, Wq_h, bq, nullptr, nullptr, q_h, ROWS, DIM, DIM);
    gemm_f16_wmma<0, false><<<gD, 256, 0, stream>>>(vn_h, Wv_h, bv, nullptr, vmap, nullptr, ROWS, DIM, DIM);

    // offsets + attention weights
    head_mlp<<<(ROWS * NHEAD + 127) / 128, 128, 0, stream>>>(q_h, Wo1, bo1, Wo2, bo2, Wa, ba, offv, attn);

    // bilinear deformable sampling + head combine
    deform_gather<<<(BSZ * NHEAD * SEQ) * 32 / 256, 256, 0, stream>>>(vmap, offv, attn, ho_h);

    // srcres = source + ho @ Wout + bout
    gemm_f16_wmma<0, false><<<gD, 256, 0, stream>>>(ho_h, Wout_h, bout, source, srcres, nullptr, ROWS, DIM, DIM);

    // FFN: ln -> gelu(x@W1+b1) -> out = srcres + h1@W2+b2
    layernorm_f16<<<ROWS, 256, 0, stream>>>(srcres, gffn, bffn_ln, ffn_h);
    gemm_f16_wmma<1, true ><<<g4, 256, 0, stream>>>(ffn_h, W1_h, b1, nullptr, nullptr, h1_h, ROWS, D4, DIM);
    gemm_f16_wmma<0, false><<<gD, 256, 0, stream>>>(h1_h, W2_h, b2, srcres, (float*)d_out, nullptr, ROWS, DIM, D4);
}
